// Attention_71030169141357
// MI455X (gfx1250) — matrix operand order repaired and checked by reading, not compiled, not run on hardware
//
#include <hip/hip_runtime.h>
#include <hip/hip_bf16.h>

#define N_EMBD 512
#define NH     8
#define NKV    2
#define HD     64
#define TSEQ   4096
#define BB     2

typedef __bf16 bf16_t;
typedef __attribute__((ext_vector_type(16))) __bf16 v16bf;
typedef __attribute__((ext_vector_type(8)))  __bf16 v8bf;
typedef __attribute__((ext_vector_type(8)))  float  v8f;

static __device__ __forceinline__ bf16_t f2bf(float f) {
  union { float f; unsigned u; } x; x.f = f;
  unsigned r = x.u + 0x7FFFu + ((x.u >> 16) & 1u);   // round-to-nearest-even
  unsigned short h = (unsigned short)(r >> 16);
  return __builtin_bit_cast(bf16_t, h);
}

static __device__ __forceinline__ v8f wmma_bf16(v16bf a, v16bf b, v8f c) {
  // D = A(16x32 bf16) * B(32x16 bf16) + C(16x16 f32)
  return __builtin_amdgcn_wmma_f32_16x16x32_bf16(false, a, false, b, (short)0, c, false, false);
}

// A-fragment (16x32 bf16), source row-major with leading dim `ld`.
// ISA layout: lane m=l%16; K elems: e[0..7]=K co..co+7, e[8..15]=K 16+co.., co=(l<16?0:8)
static __device__ __forceinline__ v16bf load_a(const bf16_t* base, int ld, int k0, int lane) {
  const int row = lane & 15;
  const int co  = (lane < 16) ? 0 : 8;
  union { v16bf v; v8bf h[2]; } u;
  u.h[0] = *(const v8bf*)(base + (size_t)row * ld + k0 + co);
  u.h[1] = *(const v8bf*)(base + (size_t)row * ld + k0 + 16 + co);
  return u.v;
}

// B-fragment (32x16 bf16) from a column-major-stored (i.e. transposed, row=n) matrix:
// lane n=l%16; K elems: e[0..7]=K co..co+7, e[8..15]=K 16+co.., co=(l<16?0:8)
static __device__ __forceinline__ v16bf load_b(const bf16_t* base, int ld, int k0, int lane) {
  const int n  = lane & 15;
  const int co = (lane < 16) ? 0 : 8;
  union { v16bf v; v8bf h[2]; } u;
  u.h[0] = *(const v8bf*)(base + (size_t)n * ld + k0 + co);
  u.h[1] = *(const v8bf*)(base + (size_t)n * ld + k0 + 16 + co);
  return u.v;
}

// ---------------- prep kernels ----------------
__global__ void cvt_x_kernel(const float* __restrict__ x, bf16_t* __restrict__ xb, int n) {
  int i = blockIdx.x * blockDim.x + threadIdx.x;
  if (i < n) xb[i] = f2bf(x[i]);
}

__global__ void build_wt_kernel(const float* __restrict__ Wq, const float* __restrict__ Wk,
                                const float* __restrict__ Wv, bf16_t* __restrict__ Wt) {
  int i = blockIdx.x * blockDim.x + threadIdx.x;  // 768*512
  int n = i >> 9, k = i & 511;
  float v = (n < 512) ? Wq[(size_t)k * 512 + n]
          : (n < 640) ? Wk[(size_t)k * 128 + (n - 512)]
                      : Wv[(size_t)k * 128 + (n - 640)];
  Wt[i] = f2bf(v);
}

__global__ void build_wot_kernel(const float* __restrict__ Wo, bf16_t* __restrict__ Wot) {
  int i = blockIdx.x * blockDim.x + threadIdx.x;  // 512*512
  int n = i >> 9, k = i & 511;
  Wot[i] = f2bf(Wo[(size_t)k * 512 + n]);
}

// ---------------- QKV projection + bias + RoPE ----------------
__global__ void __launch_bounds__(128) qkv_rope_kernel(
    const bf16_t* __restrict__ xb, const bf16_t* __restrict__ Wt,
    const float* __restrict__ bq, const float* __restrict__ bk, const float* __restrict__ bv,
    bf16_t* __restrict__ Qb, bf16_t* __restrict__ Kb, bf16_t* __restrict__ Vt) {
  const int lane  = threadIdx.x & 31;
  const int wave  = threadIdx.x >> 5;
  const int ntile = blockIdx.x;                 // 0..11 (64 output cols each)
  const int mglob = blockIdx.y * 4 + wave;      // 16-row tile over B*T
  const int b     = mglob >> 8;                 // 256 tiles per batch
  const int t0    = (mglob & 255) << 4;
  const int half8 = (lane >> 4) << 3;
  const int nidx  = lane & 15;

  const bf16_t* xrow = xb + (size_t)(b * TSEQ + t0) * N_EMBD;
  v8f acc[4] = {};
  for (int k0 = 0; k0 < N_EMBD; k0 += 32) {
    v16bf a = load_a(xrow, N_EMBD, k0, lane);
#pragma unroll
    for (int j = 0; j < 4; ++j) {
      v16bf w = load_b(Wt + (size_t)(ntile * 64 + j * 16) * N_EMBD, N_EMBD, k0, lane);
      acc[j] = wmma_bf16(a, w, acc[j]);
    }
  }
#pragma unroll
  for (int j = 0; j < 4; ++j) {
    int n = ntile * 64 + j * 16 + nidx;
    float bias = (n < 512) ? bq[n] : (n < 640 ? bk[n - 512] : bv[n - 640]);
#pragma unroll
    for (int r = 0; r < 8; ++r) acc[j][r] += bias;
  }
  if (ntile < 10) {  // RoPE on Q and K tiles (64-col tiles are head-aligned)
#pragma unroll
    for (int pj = 0; pj < 2; ++pj) {
      int d = pj * 16 + nidx;                                        // 0..31
      float invf = __expf(-(float)d * (13.815510557964274f / 32.0f)); // base^(-d/32)
#pragma unroll
      for (int r = 0; r < 8; ++r) {
        float t  = (float)(t0 + r + half8);
        float an = t * invf;
        float cs = __cosf(an), sn = __sinf(an);
        float lo = acc[pj][r], hi = acc[pj + 2][r];
        acc[pj][r]     = lo * cs - hi * sn;
        acc[pj + 2][r] = hi * cs + lo * sn;
      }
    }
  }
  if (ntile < 8) {                                   // Q: [B,H,T,D] row-major
    bf16_t* dst = Qb + (size_t)(b * NH + ntile) * TSEQ * HD;
#pragma unroll
    for (int j = 0; j < 4; ++j)
#pragma unroll
      for (int r = 0; r < 8; ++r)
        dst[(size_t)(t0 + r + half8) * HD + j * 16 + nidx] = f2bf(acc[j][r]);
  } else if (ntile < 10) {                           // K: [B,Hkv,T,D]
    bf16_t* dst = Kb + (size_t)(b * NKV + (ntile - 8)) * TSEQ * HD;
#pragma unroll
    for (int j = 0; j < 4; ++j)
#pragma unroll
      for (int r = 0; r < 8; ++r)
        dst[(size_t)(t0 + r + half8) * HD + j * 16 + nidx] = f2bf(acc[j][r]);
  } else {                                           // V transposed: [B,Hkv,D,T]
    bf16_t* dst = Vt + (size_t)(b * NKV + (ntile - 10)) * HD * TSEQ;
#pragma unroll
    for (int j = 0; j < 4; ++j)
#pragma unroll
      for (int r = 0; r < 8; ++r)
        dst[(size_t)(j * 16 + nidx) * TSEQ + (t0 + r + half8)] = f2bf(acc[j][r]);
  }
}

// ---------------- flash attention (online softmax, causal, GQA) ----------------
__global__ void __launch_bounds__(128) flash_attn_kernel(
    const bf16_t* __restrict__ Qb, const bf16_t* __restrict__ Kb,
    const bf16_t* __restrict__ Vt, bf16_t* __restrict__ Ob) {
  __shared__ float lds[4][16 * 32];
  const int lane  = threadIdx.x & 31;
  const int wave  = threadIdx.x >> 5;
  const int h     = blockIdx.y & 7;
  const int b     = blockIdx.y >> 3;
  const int kv    = h >> 2;                      // repeat(k,4) -> kv head = h/4
  const int m0    = (blockIdx.x * 4 + wave) * 16;
  const int half8 = (lane >> 4) << 3;
  const int nidx  = lane & 15;

  const bf16_t* Qt    = Qb + ((size_t)(b * NH + h) * TSEQ + m0) * HD;
  const bf16_t* Kbase = Kb + (size_t)(b * NKV + kv) * TSEQ * HD;
  const bf16_t* Vbase = Vt + (size_t)(b * NKV + kv) * HD * TSEQ;

  v16bf aQ0 = load_a(Qt, HD, 0, lane);
  v16bf aQ1 = load_a(Qt, HD, 32, lane);

  float mrow[8], lrow[8];
#pragma unroll
  for (int r = 0; r < 8; ++r) { mrow[r] = -3.0e38f; lrow[r] = 0.0f; }
  v8f accO[4] = {};
  float* P = lds[wave];
  const int tmax = m0 + 15;

  for (int s0 = 0; s0 <= tmax; s0 += 32) {
    v8f S0 = {}, S1 = {};
    {
      const bf16_t* k0p = Kbase + (size_t)s0 * HD;
      S0 = wmma_bf16(aQ0, load_b(k0p, HD, 0, lane), S0);
      S0 = wmma_bf16(aQ1, load_b(k0p, HD, 32, lane), S0);
      const bf16_t* k1p = Kbase + (size_t)(s0 + 16) * HD;
      S1 = wmma_bf16(aQ0, load_b(k1p, HD, 0, lane), S1);
      S1 = wmma_bf16(aQ1, load_b(k1p, HD, 32, lane), S1);
    }
    const bool needMask = (s0 + 31) > m0;
#pragma unroll
    for (int r = 0; r < 8; ++r) {
      int t = m0 + r + half8;
      float v0 = S0[r] * 0.125f;                 // 1/sqrt(64)
      float v1 = S1[r] * 0.125f;
      if (needMask) {
        if (s0 + nidx > t)      v0 = -3.0e38f;
        if (s0 + 16 + nidx > t) v1 = -3.0e38f;
      }
      S0[r] = v0; S1[r] = v1;
    }
    float alpha[8];
#pragma unroll
    for (int r = 0; r < 8; ++r) {                // row max within 16-lane half
      float v = fmaxf(S0[r], S1[r]);
      v = fmaxf(v, __shfl_xor(v, 1));
      v = fmaxf(v, __shfl_xor(v, 2));
      v = fmaxf(v, __shfl_xor(v, 4));
      v = fmaxf(v, __shfl_xor(v, 8));
      float mn = fmaxf(mrow[r], v);
      alpha[r] = __expf(mrow[r] - mn);
      mrow[r]  = mn;
    }
#pragma unroll
    for (int r = 0; r < 8; ++r) {
      S0[r] = __expf(S0[r] - mrow[r]);
      S1[r] = __expf(S1[r] - mrow[r]);
      float v = S0[r] + S1[r];
      v += __shfl_xor(v, 1);
      v += __shfl_xor(v, 2);
      v += __shfl_xor(v, 4);
      v += __shfl_xor(v, 8);
      lrow[r] = lrow[r] * alpha[r] + v;
    }
#pragma unroll
    for (int j = 0; j < 4; ++j)
#pragma unroll
      for (int r = 0; r < 8; ++r) accO[j][r] *= alpha[r];

    // stage P (C-layout) into LDS as row-major [16][32]
#pragma unroll
    for (int r = 0; r < 8; ++r) {
      P[(r + half8) * 32 + nidx]      = S0[r];
      P[(r + half8) * 32 + 16 + nidx] = S1[r];
    }
    asm volatile("s_wait_dscnt 0x0" ::: "memory");
    // gather A-fragment of P (16x32), convert to bf16
    union { v16bf v; bf16_t e[16]; } uP;
    {
      const int mr = lane & 15;
      const int c0 = (lane < 16) ? 0 : 8;
#pragma unroll
      for (int i = 0; i < 8; ++i) uP.e[i]     = f2bf(P[mr * 32 + c0 + i]);
#pragma unroll
      for (int i = 0; i < 8; ++i) uP.e[8 + i] = f2bf(P[mr * 32 + 16 + c0 + i]);
    }
    asm volatile("s_wait_dscnt 0x0" ::: "memory");
    // O += P * V   (B from V^T layout [D][T]: contiguous in s)
#pragma unroll
    for (int j = 0; j < 4; ++j) {
      v16bf bv = load_b(Vbase + (size_t)(j * 16) * TSEQ, TSEQ, s0, lane);
      accO[j] = wmma_bf16(uP.v, bv, accO[j]);
    }
  }
  // epilogue: normalize and store to Ob[B,T, h*64+d] (bf16 for out-proj GEMM)
  bf16_t* dst = Ob + (size_t)b * TSEQ * N_EMBD + (size_t)h * HD;
#pragma unroll
  for (int r = 0; r < 8; ++r) {
    float inv = 1.0f / lrow[r];
    int t = m0 + r + half8;
#pragma unroll
    for (int j = 0; j < 4; ++j)
      dst[(size_t)t * N_EMBD + j * 16 + nidx] = f2bf(accO[j][r] * inv);
  }
}

// ---------------- output projection ----------------
__global__ void __launch_bounds__(128) out_proj_kernel(
    const bf16_t* __restrict__ Ob, const bf16_t* __restrict__ Wot,
    float* __restrict__ out) {
  const int lane  = threadIdx.x & 31;
  const int wave  = threadIdx.x >> 5;
  const int ntile = blockIdx.x;                 // 0..7
  const int mglob = blockIdx.y * 4 + wave;
  const bf16_t* orow = Ob + (size_t)mglob * 16 * N_EMBD;
  v8f acc[4] = {};
  for (int k0 = 0; k0 < N_EMBD; k0 += 32) {
    v16bf a = load_a(orow, N_EMBD, k0, lane);
#pragma unroll
    for (int j = 0; j < 4; ++j) {
      v16bf w = load_b(Wot + (size_t)(ntile * 64 + j * 16) * N_EMBD, N_EMBD, k0, lane);
      acc[j] = wmma_bf16(a, w, acc[j]);
    }
  }
  const int half8 = (lane >> 4) << 3;
  const int nidx  = lane & 15;
#pragma unroll
  for (int j = 0; j < 4; ++j)
#pragma unroll
    for (int r = 0; r < 8; ++r) {
      size_t row = (size_t)mglob * 16 + r + half8;
      out[row * N_EMBD + ntile * 64 + j * 16 + nidx] = acc[j][r];
    }
}

extern "C" void kernel_launch(void* const* d_in, const int* in_sizes, int n_in,
                              void* d_out, int out_size, void* d_ws, size_t ws_size,
                              hipStream_t stream) {
  (void)in_sizes; (void)n_in; (void)out_size; (void)ws_size;
  const float* x  = (const float*)d_in[0];
  const float* Wq = (const float*)d_in[1];
  const float* bq = (const float*)d_in[2];
  const float* Wk = (const float*)d_in[3];
  const float* bk = (const float*)d_in[4];
  const float* Wv = (const float*)d_in[5];
  const float* bv = (const float*)d_in[6];
  const float* Wo = (const float*)d_in[7];
  float* out = (float*)d_out;

  char* ws = (char*)d_ws;
  size_t off = 0;
  auto take = [&](size_t bytes) {
    char* p = ws + off;
    off += (bytes + 255) & ~(size_t)255;
    return p;
  };
  bf16_t* xb  = (bf16_t*)take((size_t)BB * TSEQ * N_EMBD * 2);
  bf16_t* Wt  = (bf16_t*)take((size_t)768 * 512 * 2);
  bf16_t* Wot = (bf16_t*)take((size_t)512 * 512 * 2);
  bf16_t* Qb  = (bf16_t*)take((size_t)BB * NH * TSEQ * HD * 2);
  bf16_t* Kb  = (bf16_t*)take((size_t)BB * NKV * TSEQ * HD * 2);
  bf16_t* Vt  = (bf16_t*)take((size_t)BB * NKV * HD * TSEQ * 2);
  bf16_t* Ob  = (bf16_t*)take((size_t)BB * TSEQ * N_EMBD * 2);

  { int n = BB * TSEQ * N_EMBD;
    cvt_x_kernel<<<(n + 255) / 256, 256, 0, stream>>>(x, xb, n); }
  build_wt_kernel<<<(768 * 512) / 256, 256, 0, stream>>>(Wq, Wk, Wv, Wt);
  build_wot_kernel<<<(512 * 512) / 256, 256, 0, stream>>>(Wo, Wot);
  qkv_rope_kernel<<<dim3(12, BB * TSEQ / 64), 128, 0, stream>>>(xb, Wt, bq, bk, bv, Qb, Kb, Vt);
  flash_attn_kernel<<<dim3(TSEQ / 64, BB * NH), 128, 0, stream>>>(Qb, Kb, Vt, Ob);
  out_proj_kernel<<<dim3(8, BB * TSEQ / 64), 128, 0, stream>>>(Ob, Wot, out);
}
